// Transformer_Based_Model_51092930953794
// MI455X (gfx1250) — matrix operand order repaired and checked by reading, not compiled, not run on hardware
//
#include <hip/hip_runtime.h>
#include <hip/hip_bf16.h>

// ---------------------------------------------------------------------------
// MI455X / gfx1250 implementation.
// Dense algebra via v_wmma_f32_16x16x32_bf16 (wave32), double-buffered
// LDS-staged bf16 tiles: b128 global loads overlap WMMA, single barrier/step.
// ---------------------------------------------------------------------------

constexpr int Bc   = 4;
constexpr int Lc   = 1024;
constexpr int Dc   = 1024;
constexpr int Hc   = 16;
constexpr int HDc  = 64;
constexpr int DFFc = 4096;
constexpr int NLc  = 4;

typedef __attribute__((ext_vector_type(16))) __bf16 v16bf;
typedef __attribute__((ext_vector_type(8)))  __bf16 v8bf;
typedef __attribute__((ext_vector_type(4)))  __bf16 v4bf;
typedef __attribute__((ext_vector_type(8)))  float  v8f;

__device__ __forceinline__ __bf16 f2bf(float f) {
  union { float f; unsigned u; } v; v.f = f;
  unsigned r = (v.u + 0x7FFFu + ((v.u >> 16) & 1u)) >> 16;   // round-to-nearest-even
  unsigned short s = (unsigned short)r;
  __bf16 b; __builtin_memcpy(&b, &s, 2);
  return b;
}

__device__ __forceinline__ v4bf cvt4(float4 f) {
  v4bf r;
  r[0] = f2bf(f.x); r[1] = f2bf(f.y); r[2] = f2bf(f.z); r[3] = f2bf(f.w);
  return r;
}

__device__ __forceinline__ v16bf cat16(v8bf lo, v8bf hi) {
  return __builtin_shufflevector(lo, hi, 0, 1, 2, 3, 4, 5, 6, 7,
                                 8, 9, 10, 11, 12, 13, 14, 15);
}

// ---------------------------------------------------------------------------
// Generic batched GEMM: C[z] = epi(alpha * A[z] @ W[z] + bias)
//   A: M x K (row major, lda); W: K x N (row major, ldw) or TRANSW -> W[n*ldw+k]
//   Block: 256 threads = 8 waves, block tile 64x64 (4 waves M x 2 waves N).
//   Each wave computes a 16x32 strip: one A fragment feeds 2 WMMAs per k-step.
//   Double-buffered LDS: global b128 loads for tile i+1 issued before the
//   WMMAs on tile i; one __syncthreads per k-step.
//   Requires M%64==0, N%64==0, K%32==0 (true for every call site here).
// epi: 0=none 1=sigmoid 2=tanh 3=gelu(tanh)
// ---------------------------------------------------------------------------
#define LDSP 40   // row stride in bf16 (80B, 16B aligned, spreads banks)

template <int TRANSW>
__device__ __forceinline__ void load_tile(const float* __restrict__ Ab,
                                          const float* __restrict__ Wb,
                                          int lda, int ldw, int m0, int n0,
                                          int r0, int c0, int k0,
                                          float4& a0, float4& a1,
                                          float4& w0, float4& w1)
{
  a0 = *(const float4*)&Ab[(size_t)(m0 + r0)      * lda + (k0 + c0)];
  a1 = *(const float4*)&Ab[(size_t)(m0 + r0 + 32) * lda + (k0 + c0)];
  if (TRANSW) {  // W element (k,n) = Wb[n*ldw + k]: k-contiguous -> b128
    w0 = *(const float4*)&Wb[(size_t)(n0 + r0)      * ldw + (k0 + c0)];
    w1 = *(const float4*)&Wb[(size_t)(n0 + r0 + 32) * ldw + (k0 + c0)];
  } else {       // W element (k,n) = Wb[k*ldw + n]: gather 4 strided k
    const float* wp0 = &Wb[(size_t)(k0 + c0) * ldw + (n0 + r0)];
    const float* wp1 = wp0 + 32;
    w0.x = wp0[0]; w0.y = wp0[(size_t)ldw]; w0.z = wp0[2 * (size_t)ldw]; w0.w = wp0[3 * (size_t)ldw];
    w1.x = wp1[0]; w1.y = wp1[(size_t)ldw]; w1.z = wp1[2 * (size_t)ldw]; w1.w = wp1[3 * (size_t)ldw];
  }
}

template <int TRANSW>
__global__ __launch_bounds__(256)
void gemm_bf16_wmma(const float* __restrict__ A, int lda, long long sA,
                    const float* __restrict__ Wt, int ldw, long long sW,
                    const float* __restrict__ bias,
                    float* __restrict__ C, int ldc, long long sC,
                    int M, int N, int K, float alpha, int epi)
{
  __shared__ __bf16 As[2][64][LDSP];   // [buf][m][k]
  __shared__ __bf16 Ws[2][64][LDSP];   // [buf][n][k] (k-contiguous per n)

  const int tid  = threadIdx.x;
  const int lane = tid & 31;
  const int wave = tid >> 5;
  const int wm   = wave & 3;     // wave row (4 along M)
  const int wn   = wave >> 2;    // wave col (2 along N, 32 cols each)
  const int half = lane >> 4;    // ISA wave32 lane-half
  const int l16  = lane & 15;

  const long long z = blockIdx.z;
  const float* Ab = A  + z * sA;
  const float* Wb = Wt + z * sW;
  float*       Cb = C  + z * sC;

  const int m0 = blockIdx.y * 64;
  const int n0 = blockIdx.x * 64;

  // Staging assignment: quad (4 f32 -> 4 bf16) granularity.
  const int r0 = tid >> 3;          // 0..31 (row / n within tile; +32 second half)
  const int c0 = (tid & 7) * 4;     // 0..28 (k within tile)

  v8f acc0 = {0.f, 0.f, 0.f, 0.f, 0.f, 0.f, 0.f, 0.f};
  v8f acc1 = {0.f, 0.f, 0.f, 0.f, 0.f, 0.f, 0.f, 0.f};

  // ---- prologue: stage tile 0
  float4 a0, a1, w0, w1;
  load_tile<TRANSW>(Ab, Wb, lda, ldw, m0, n0, r0, c0, 0, a0, a1, w0, w1);
  *(v4bf*)&As[0][r0][c0]      = cvt4(a0);
  *(v4bf*)&As[0][r0 + 32][c0] = cvt4(a1);
  *(v4bf*)&Ws[0][r0][c0]      = cvt4(w0);
  *(v4bf*)&Ws[0][r0 + 32][c0] = cvt4(w1);
  __syncthreads();

  int buf = 0;
  for (int k0 = 0; k0 < K; k0 += 32) {
    const bool more = (k0 + 32) < K;
    if (more) {  // issue next tile's global loads; in flight across the WMMAs
      load_tile<TRANSW>(Ab, Wb, lda, ldw, m0, n0, r0, c0, k0 + 32, a0, a1, w0, w1);
      __builtin_prefetch(&Ab[(size_t)(m0 + (tid & 63)) * lda + (k0 + 64)], 0, 1);
    }

    // ---- fragments from current buffer: contiguous 16B ds_load_b128 pairs
    // A 16x32: lane half -> K = half*16 + e, contiguous 16 elems in As[m][k]
    const __bf16* arow = &As[buf][wm * 16 + l16][0];
    v16bf af = cat16(*(const v8bf*)&arow[half * 16],
                     *(const v8bf*)&arow[half * 16 + 8]);
    // B 32x16: lane half -> K = half*16 + e, contiguous 16 elems in Ws[n][k]
    const __bf16* brow0 = &Ws[buf][wn * 32 + l16][0];
    const __bf16* brow1 = &Ws[buf][wn * 32 + 16 + l16][0];
    v16bf bf0 = cat16(*(const v8bf*)&brow0[half * 16],
                      *(const v8bf*)&brow0[half * 16 + 8]);
    v16bf bf1 = cat16(*(const v8bf*)&brow1[half * 16],
                      *(const v8bf*)&brow1[half * 16 + 8]);

    acc0 = __builtin_amdgcn_wmma_f32_16x16x32_bf16(false, af, false, bf0,
                                                   (short)0, acc0, false, false);
    acc1 = __builtin_amdgcn_wmma_f32_16x16x32_bf16(false, af, false, bf1,
                                                   (short)0, acc1, false, false);

    if (more) {  // convert + stage next tile into the other buffer
      const int nb = buf ^ 1;
      *(v4bf*)&As[nb][r0][c0]      = cvt4(a0);
      *(v4bf*)&As[nb][r0 + 32][c0] = cvt4(a1);
      *(v4bf*)&Ws[nb][r0][c0]      = cvt4(w0);
      *(v4bf*)&Ws[nb][r0 + 32][c0] = cvt4(w1);
    }
    __syncthreads();
    buf ^= 1;
  }

  // ---- epilogue: C/D layout VGPR j -> M = half*8 + j, N = l16
  const int row0 = m0 + wm * 16 + half * 8;
  const int colA = n0 + wn * 32 + l16;
  const int colB = colA + 16;
#pragma unroll
  for (int j = 0; j < 8; ++j) {
    float va = acc0[j] * alpha;
    float vb = acc1[j] * alpha;
    if (bias) { va += bias[colA]; vb += bias[colB]; }
    if (epi == 1) {
      va = 1.f / (1.f + __expf(-va)); vb = 1.f / (1.f + __expf(-vb));
    } else if (epi == 2) {
      va = tanhf(va); vb = tanhf(vb);
    } else if (epi == 3) {
      float a3 = va * va * va, b3 = vb * vb * vb;
      va = 0.5f * va * (1.f + tanhf(0.7978845608f * (va + 0.044715f * a3)));
      vb = 0.5f * vb * (1.f + tanhf(0.7978845608f * (vb + 0.044715f * b3)));
    }
    Cb[(size_t)(row0 + j) * ldc + colA] = va;
    Cb[(size_t)(row0 + j) * ldc + colB] = vb;
  }
}

// ---------------------------------------------------------------------------
// LayerNorm over rows of width Wd (256 threads/row, regs-cached, in-place safe)
// input = x + resScale*res (res nullable)
// ---------------------------------------------------------------------------
__global__ __launch_bounds__(256)
void ln_kernel(const float* __restrict__ x, const float* __restrict__ res,
               float resScale, const float* __restrict__ g,
               const float* __restrict__ b, float* __restrict__ out,
               int Wd, float eps)
{
  __shared__ float red[256];
  const int tid = threadIdx.x;
  const size_t base = (size_t)blockIdx.x * Wd;
  const int ne = Wd >> 8;  // 4 (Wd=1024) or 8 (Wd=2048)
  float v[8];
  float s = 0.f;
  for (int k = 0; k < ne; ++k) {
    int j = tid + (k << 8);
    float t = x[base + j];
    if (res) t += resScale * res[base + j];
    v[k] = t; s += t;
  }
  red[tid] = s; __syncthreads();
  for (int o = 128; o > 0; o >>= 1) { if (tid < o) red[tid] += red[tid + o]; __syncthreads(); }
  float mu = red[0] / Wd; __syncthreads();
  float q = 0.f;
  for (int k = 0; k < ne; ++k) { float d = v[k] - mu; q += d * d; }
  red[tid] = q; __syncthreads();
  for (int o = 128; o > 0; o >>= 1) { if (tid < o) red[tid] += red[tid + o]; __syncthreads(); }
  float rstd = rsqrtf(red[0] / Wd + eps); __syncthreads();
  for (int k = 0; k < ne; ++k) {
    int j = tid + (k << 8);
    out[base + j] = (v[k] - mu) * rstd * g[j] + b[j];
  }
}

// ---------------------------------------------------------------------------
// Differential softmax: in-place on S rows of length L.
// row = (b*H + h)*L + q.  p1 = softmax(mask(S[j])), p2 = softmax(mask(S[(j-1)%L]))
// S[j] <- lam * (p1[j] - p2[j])
// ---------------------------------------------------------------------------
__global__ __launch_bounds__(256)
void softmax_diff(float* __restrict__ S, const int* __restrict__ mask,
                  const float* __restrict__ lamp)
{
  __shared__ float red[256];
  const int tid = threadIdx.x;
  const long long row = blockIdx.x;
  const int b = (int)(row / ((long long)Hc * Lc));
  float* s = S + row * (long long)Lc;
  const int* mrow = mask + (size_t)b * Lc;
  const float lam = lamp[0];

  float v1[4], v2[4];
  for (int k = 0; k < 4; ++k) {
    int j = tid + (k << 8);
    float sv = s[j];
    float sp = s[(j + Lc - 1) & (Lc - 1)];   // rolled score (kp = roll(k,1))
    bool ok = mrow[j] != 0;
    v1[k] = ok ? sv : -1.0e9f;
    v2[k] = ok ? sp : -1.0e9f;
  }
  float m = fmaxf(fmaxf(v1[0], v1[1]), fmaxf(v1[2], v1[3]));
  red[tid] = m; __syncthreads();
  for (int o = 128; o > 0; o >>= 1) { if (tid < o) red[tid] = fmaxf(red[tid], red[tid + o]); __syncthreads(); }
  float m1 = red[0]; __syncthreads();
  m = fmaxf(fmaxf(v2[0], v2[1]), fmaxf(v2[2], v2[3]));
  red[tid] = m; __syncthreads();
  for (int o = 128; o > 0; o >>= 1) { if (tid < o) red[tid] = fmaxf(red[tid], red[tid + o]); __syncthreads(); }
  float m2 = red[0]; __syncthreads();

  float s1 = 0.f, s2 = 0.f;
  for (int k = 0; k < 4; ++k) { s1 += __expf(v1[k] - m1); s2 += __expf(v2[k] - m2); }
  red[tid] = s1; __syncthreads();
  for (int o = 128; o > 0; o >>= 1) { if (tid < o) red[tid] += red[tid + o]; __syncthreads(); }
  float S1r = red[0]; __syncthreads();
  red[tid] = s2; __syncthreads();
  for (int o = 128; o > 0; o >>= 1) { if (tid < o) red[tid] += red[tid + o]; __syncthreads(); }
  float S2r = red[0]; __syncthreads();

  float r1 = 1.f / S1r, r2 = 1.f / S2r;
  for (int k = 0; k < 4; ++k) {
    int j = tid + (k << 8);
    s[j] = lam * (__expf(v1[k] - m1) * r1 - __expf(v2[k] - m2) * r2);
  }
}

// ---------------------------------------------------------------------------
// Elementwise / reshuffle kernels
// ---------------------------------------------------------------------------
__global__ void rope_kernel(const float* __restrict__ x, float* __restrict__ xp)
{
  size_t idx = (size_t)blockIdx.x * blockDim.x + threadIdx.x;
  const size_t total = (size_t)Bc * Lc * (Dc / 2);
  if (idx >= total) return;
  int i = (int)(idx % (Dc / 2));
  int l = (int)((idx / (Dc / 2)) % Lc);
  size_t base = (idx / (Dc / 2)) * Dc + 2 * (size_t)i;
  float inv = powf(10000.f, -2.f * (float)i / (float)Dc);
  float f = (float)l * inv;
  float c = cosf(f), sn = sinf(f);
  float x1 = x[base], x2 = x[base + 1];
  xp[base]     = x1 * c - x2 * sn;
  xp[base + 1] = x1 * sn + x2 * c;
}

__global__ void conv_dw3(const float* __restrict__ x, const float* __restrict__ w,
                         const float* __restrict__ bias, float* __restrict__ out)
{
  size_t idx = (size_t)blockIdx.x * blockDim.x + threadIdx.x;
  if (idx >= (size_t)Bc * Lc * Dc) return;
  int d = (int)(idx % Dc);
  int l = (int)((idx / Dc) % Lc);
  float xm = (l > 0)      ? x[idx - Dc] : 0.f;
  float x0 = x[idx];
  float xq = (l < Lc - 1) ? x[idx + Dc] : 0.f;
  out[idx] = xm * w[d * 3 + 0] + x0 * w[d * 3 + 1] + xq * w[d * 3 + 2] + bias[d];
}

__global__ void delta_mix(const float* __restrict__ x, const float* __restrict__ gate,
                          const float* __restrict__ delta, const float* __restrict__ alphap,
                          float* __restrict__ out)
{
  size_t idx = (size_t)blockIdx.x * blockDim.x + threadIdx.x;
  if (idx >= (size_t)Bc * Lc * Dc) return;
  out[idx] = x[idx] + alphap[0] * gate[idx] * delta[idx];
}

__global__ void concat2(const float* __restrict__ a, const float* __restrict__ b,
                        float* __restrict__ out)
{
  size_t idx = (size_t)blockIdx.x * blockDim.x + threadIdx.x;
  if (idx >= (size_t)Bc * Lc * 2 * Dc) return;
  int c = (int)(idx % (2 * Dc));
  size_t r = idx / (2 * Dc);
  out[idx] = (c < Dc) ? a[r * Dc + c] : b[r * Dc + (c - Dc)];
}

__global__ __launch_bounds__(256)
void rowdot_sigmoid(const float* __restrict__ A, const float* __restrict__ w,
                    const float* __restrict__ b, float* __restrict__ out, int Kd)
{
  int lane = threadIdx.x & 31;
  int wave = threadIdx.x >> 5;
  int row = blockIdx.x * 8 + wave;
  const float* a = A + (size_t)row * Kd;
  float s = 0.f;
  for (int k = lane; k < Kd; k += 32) s += a[k] * w[k];
#pragma unroll
  for (int o = 16; o > 0; o >>= 1) s += __shfl_xor(s, o, 32);
  if (lane == 0) out[row] = 1.f / (1.f + __expf(-(s + b[0])));
}

__global__ void gate_mix(const float* __restrict__ g, const float* __restrict__ dx,
                         const float* __restrict__ xp, float* __restrict__ out)
{
  size_t idx = (size_t)blockIdx.x * blockDim.x + threadIdx.x;
  if (idx >= (size_t)Bc * Lc * Dc) return;
  float gv = g[idx / Dc];
  out[idx] = gv * dx[idx] + (1.f - gv) * xp[idx];
}

__global__ void split_heads(const float* __restrict__ qkv, float* __restrict__ q,
                            float* __restrict__ k, float* __restrict__ v)
{
  size_t idx = (size_t)blockIdx.x * blockDim.x + threadIdx.x;
  if (idx >= (size_t)Bc * Lc * Dc) return;
  int d = (int)(idx % HDc);
  int h = (int)((idx / HDc) % Hc);
  size_t bl = idx / Dc;
  int l = (int)(bl % Lc);
  int b = (int)(bl / Lc);
  size_t src = bl * (size_t)(3 * Dc) + (size_t)h * HDc + d;
  size_t dst = (((size_t)b * Hc + h) * Lc + l) * HDc + d;
  q[dst] = qkv[src];
  k[dst] = qkv[src + Dc];
  v[dst] = qkv[src + 2 * Dc];
}

__global__ void merge_heads(const float* __restrict__ o, float* __restrict__ out)
{
  size_t idx = (size_t)blockIdx.x * blockDim.x + threadIdx.x;
  if (idx >= (size_t)Bc * Lc * Dc) return;
  int d = (int)(idx % HDc);
  int h = (int)((idx / HDc) % Hc);
  size_t bl = idx / Dc;
  int l = (int)(bl % Lc);
  int b = (int)(bl / Lc);
  out[idx] = o[(((size_t)b * Hc + h) * Lc + l) * HDc + d];
}

__global__ void final_out(const float* __restrict__ xp, const float* __restrict__ pd,
                          float* __restrict__ out)
{
  size_t idx = (size_t)blockIdx.x * blockDim.x + threadIdx.x;
  if (idx >= (size_t)Bc * Lc * Dc) return;
  out[idx] = xp[idx] + 0.5f * pd[idx];
}

// ---------------------------------------------------------------------------
// Host-side orchestration
// ---------------------------------------------------------------------------
extern "C" void kernel_launch(void* const* d_in, const int* in_sizes, int n_in,
                              void* d_out, int out_size, void* d_ws, size_t ws_size,
                              hipStream_t stream)
{
  (void)in_sizes; (void)n_in; (void)out_size; (void)ws_size;

  const float* x      = (const float*)d_in[0];
  const int*   mask   = (const int*)  d_in[1];
  const float* conv_w = (const float*)d_in[2];
  const float* conv_b = (const float*)d_in[3];
  const float* dln_g  = (const float*)d_in[4];
  const float* dln_b  = (const float*)d_in[5];
  const float* gate_w = (const float*)d_in[6];
  const float* gate_b = (const float*)d_in[7];
  const float* alphap = (const float*)d_in[8];
  const float* cg_g   = (const float*)d_in[9];
  const float* cg_b   = (const float*)d_in[10];
  const float* cg_w1  = (const float*)d_in[11];
  const float* cg_b1  = (const float*)d_in[12];
  const float* cg_w2  = (const float*)d_in[13];
  const float* cg_b2  = (const float*)d_in[14];
  const float* pre_g  = (const float*)d_in[15];
  const float* pre_b  = (const float*)d_in[16];
  const int LBASE = 17;   // then 15 tensors per layer

  const int BL = Bc * Lc;                 // 4096 rows
  const size_t S1 = (size_t)BL * Dc;      // one activation tensor

  float* wsf = (float*)d_ws;
  size_t off = 0;
  auto alloc = [&](size_t n) { float* p = wsf + off; off += n; return p; };
  float* xp    = alloc(S1);
  float* delta = alloc(S1);
  float* gateb = alloc(S1);
  float* dx    = alloc(S1);
  float* gi    = alloc(2 * S1);
  float* tbuf  = alloc(S1);
  float* gcoef = alloc((size_t)BL);
  float* pd    = alloc(S1);
  float* tmp   = alloc(S1);
  float* qkvb  = alloc(3 * S1);
  float* qbh   = alloc(S1);
  float* kbh   = alloc(S1);
  float* vbh   = alloc(S1);
  float* Sb    = alloc((size_t)Bc * Hc * Lc * Lc);  // attention scores / diff
  float* obht  = alloc(S1);
  float* obld  = alloc(S1);
  float* flnb  = alloc(S1);
  float* hbuf  = alloc((size_t)BL * DFFc);

  auto gemm = [&](const float* A, int lda, long long sA,
                  const float* W, int ldw, long long sW, int transW,
                  const float* bias, float* C, int ldc, long long sC,
                  int M, int N, int K, float alpha, int epi, int batches) {
    dim3 g(N / 64, M / 64, batches);
    if (transW)
      gemm_bf16_wmma<1><<<g, 256, 0, stream>>>(A, lda, sA, W, ldw, sW,
                                               bias, C, ldc, sC, M, N, K, alpha, epi);
    else
      gemm_bf16_wmma<0><<<g, 256, 0, stream>>>(A, lda, sA, W, ldw, sW,
                                               bias, C, ldc, sC, M, N, K, alpha, epi);
  };
  auto ew = [&](size_t n) { return dim3((unsigned)((n + 255) / 256)); };

  const float scale = 1.0f / sqrtf((float)Dc);   // D^-0.5 per reference

  // ---- front end ----
  rope_kernel<<<ew(S1 / 2), 256, 0, stream>>>(x, xp);
  conv_dw3<<<ew(S1), 256, 0, stream>>>(x, conv_w, conv_b, delta);
  ln_kernel<<<BL, 256, 0, stream>>>(delta, nullptr, 0.f, dln_g, dln_b, delta, Dc, 1e-5f);
  gemm(x, Dc, 0, gate_w, Dc, 0, 0, gate_b, gateb, Dc, 0, BL, Dc, Dc, 1.f, 1 /*sigmoid*/, 1);
  delta_mix<<<ew(S1), 256, 0, stream>>>(x, gateb, delta, alphap, dx);
  concat2<<<ew(2 * S1), 256, 0, stream>>>(xp, dx, gi);
  ln_kernel<<<BL, 256, 0, stream>>>(gi, nullptr, 0.f, cg_g, cg_b, gi, 2 * Dc, 1e-5f);
  gemm(gi, 2 * Dc, 0, cg_w1, Dc, 0, 0, cg_b1, tbuf, Dc, 0, BL, Dc, 2 * Dc, 1.f, 2 /*tanh*/, 1);
  rowdot_sigmoid<<<BL / 8, 256, 0, stream>>>(tbuf, cg_w2, cg_b2, gcoef, Dc);
  gate_mix<<<ew(S1), 256, 0, stream>>>(gcoef, dx, xp, tmp);
  ln_kernel<<<BL, 256, 0, stream>>>(tmp, xp, 1.f, pre_g, pre_b, pd, Dc, 1e-6f);

  // ---- layers ----
  for (int li = 0; li < NLc; ++li) {
    const float* qkv_w = (const float*)d_in[LBASE + li * 15 + 0];
    const float* qkv_b = (const float*)d_in[LBASE + li * 15 + 1];
    const float* out_w = (const float*)d_in[LBASE + li * 15 + 2];
    const float* out_b = (const float*)d_in[LBASE + li * 15 + 3];
    const float* lamp  = (const float*)d_in[LBASE + li * 15 + 4];
    const float* n1_g  = (const float*)d_in[LBASE + li * 15 + 5];
    const float* n1_b  = (const float*)d_in[LBASE + li * 15 + 6];
    const float* n2_g  = (const float*)d_in[LBASE + li * 15 + 7];
    const float* n2_b  = (const float*)d_in[LBASE + li * 15 + 8];
    const float* fln_g = (const float*)d_in[LBASE + li * 15 + 9];
    const float* fln_b = (const float*)d_in[LBASE + li * 15 + 10];
    const float* w1    = (const float*)d_in[LBASE + li * 15 + 11];
    const float* b1    = (const float*)d_in[LBASE + li * 15 + 12];
    const float* w2    = (const float*)d_in[LBASE + li * 15 + 13];
    const float* b2    = (const float*)d_in[LBASE + li * 15 + 14];

    // qkv projection
    gemm(pd, Dc, 0, qkv_w, 3 * Dc, 0, 0, qkv_b, qkvb, 3 * Dc, 0, BL, 3 * Dc, Dc, 1.f, 0, 1);
    split_heads<<<ew(S1), 256, 0, stream>>>(qkvb, qbh, kbh, vbh);

    // scores S = scale * q @ k^T, batched over B*H (a2 derived via roll in softmax_diff)
    gemm(qbh, HDc, (long long)Lc * HDc,
         kbh, HDc, (long long)Lc * HDc, 1 /*transW: k^T*/,
         nullptr, Sb, Lc, (long long)Lc * Lc,
         Lc, Lc, HDc, scale, 0, Bc * Hc);

    // lam * (softmax(S) - softmax(roll(S)))  in place
    softmax_diff<<<Bc * Hc * Lc, 256, 0, stream>>>(Sb, mask, lamp);

    // o = diff @ v, batched over B*H
    gemm(Sb, Lc, (long long)Lc * Lc,
         vbh, HDc, (long long)Lc * HDc, 0,
         nullptr, obht, HDc, (long long)Lc * HDc,
         Lc, HDc, Lc, 1.f, 0, Bc * Hc);
    merge_heads<<<ew(S1), 256, 0, stream>>>(obht, obld);

    // out projection + residual LN (n1)
    gemm(obld, Dc, 0, out_w, Dc, 0, 0, out_b, tmp, Dc, 0, BL, Dc, Dc, 1.f, 0, 1);
    ln_kernel<<<BL, 256, 0, stream>>>(tmp, pd, 1.f, n1_g, n1_b, pd, Dc, 1e-6f);

    // FFN: fln -> w1(gelu) -> w2 ; pd = ln(pd + (h@w2 + b2 + pd)) = ln(tmp + 2*pd)
    ln_kernel<<<BL, 256, 0, stream>>>(pd, nullptr, 0.f, fln_g, fln_b, flnb, Dc, 1e-6f);
    gemm(flnb, Dc, 0, w1, DFFc, 0, 0, b1, hbuf, DFFc, 0, BL, DFFc, Dc, 1.f, 3 /*gelu*/, 1);
    gemm(hbuf, DFFc, 0, w2, Dc, 0, 0, b2, tmp, Dc, 0, BL, Dc, DFFc, 1.f, 0, 1);
    ln_kernel<<<BL, 256, 0, stream>>>(tmp, pd, 2.f, n2_g, n2_b, pd, Dc, 1e-6f);
  }

  final_out<<<ew(S1), 256, 0, stream>>>(xp, pd, (float*)d_out);
}
